// LanguageIDModel_90795608637805
// MI455X (gfx1250) — compile-verified, matrix operand order checked by reading, not run on hardware
//
#include <hip/hip_runtime.h>
#include <math.h>

#define T_STEPS 128
#define BATCH   1024
#define IN_DIM  47
#define HID     646
#define OUT_DIM 5
#define KP      672   // HID padded to 21*32
#define KP0     64    // IN padded to 2*32
#define NP      704   // weight rows padded to 11*64 (no N-guard in staging)
#define LDA     40    // LDS row stride in bf16 (32 + 8 pad -> bank-spread b128 reads)
#define K2STEPS (KP / 32)   // 21

typedef __bf16 bf16;
typedef __bf16 bf16x16 __attribute__((ext_vector_type(16)));
typedef float  f32x8   __attribute__((ext_vector_type(8)));
typedef int    v4i     __attribute__((ext_vector_type(4)));

#if __has_builtin(__builtin_amdgcn_global_load_async_to_lds_b128) && \
    __has_builtin(__builtin_amdgcn_s_wait_asynccnt)
#define HAS_ASYNC_LDS 1
typedef __attribute__((address_space(1))) v4i global_v4i;
typedef __attribute__((address_space(3))) v4i local_v4i;
#else
#define HAS_ASYNC_LDS 0
#endif

union FragU { bf16x16 v; uint4 q[2]; };

__device__ inline float fast_tanh(float x) {
#if __has_builtin(__builtin_amdgcn_tanhf)
    return __builtin_amdgcn_tanhf(x);   // native v_tanh_f32 on gfx1250
#else
    return tanhf(x);
#endif
}

// 16-byte global -> LDS copy; async (ASYNCcnt) when available.
__device__ inline void cp16B(bf16* dst, const bf16* src) {
#if HAS_ASYNC_LDS
    __builtin_amdgcn_global_load_async_to_lds_b128(
        (global_v4i*)src, (local_v4i*)dst, 0, 0);
#else
    *(uint4*)dst = *(const uint4*)src;
#endif
}

// Stage one 64x32 A tile and one 64x32 W tile (4 x b128 per thread, 128 threads).
__device__ inline void stage_tiles(bf16* Asb, bf16* Wsb,
                                   const bf16* A, int ald, int mblk,
                                   const bf16* W, int wld, int nblk,
                                   int k0, int lrow, int lhalf)
{
    const bf16* asrc = A + (size_t)(mblk + lrow) * ald + k0 + lhalf * 16;
    bf16*       adst = Asb + lrow * LDA + lhalf * 16;
    cp16B(adst,     asrc);
    cp16B(adst + 8, asrc + 8);
    const bf16* wsrc = W + (size_t)(nblk + lrow) * wld + k0 + lhalf * 16;
    bf16*       wdst = Wsb + lrow * LDA + lhalf * 16;
    cp16B(wdst,     wsrc);
    cp16B(wdst + 8, wsrc + 8);
}

// Fused Elman cell tile kernel (all operands bf16, fully padded -> no guards in hot loop):
//   out[1024][KP] = tanh(A1 @ W1^T + A2 @ W2^T + b1 + b2), cols >= HID written as 0.
// A1: bf16 [1024][LD1]; W1: bf16 [NP][LD1]; A2: bf16 [1024][KP]; W2: bf16 [NP][KP].
template <int K1S, int LD1>
__global__ __launch_bounds__(128)
void rnn_cell_kernel(const bf16*  __restrict__ a1,
                     const bf16*  __restrict__ W1,
                     const bf16*  __restrict__ a2,
                     const bf16*  __restrict__ W2,
                     const float* __restrict__ b1,
                     const float* __restrict__ b2,
                     bf16*        __restrict__ out)
{
    __shared__ __align__(16) bf16 As[2][64 * LDA];
    __shared__ __align__(16) bf16 Ws[2][64 * LDA];

    const int tid   = threadIdx.x;
    const int lane  = tid & 31;
    const int wave  = tid >> 5;
    const int wm    = wave & 1;        // 2 waves along M
    const int wn    = wave >> 1;       // 2 waves along N
    const int rsel  = lane & 15;
    const int hi    = lane >> 4;
    const int c0    = hi << 3;         // K-chunk base: 0 or 8
    const int mblk  = blockIdx.y * 64;
    const int nblk  = blockIdx.x * 64;
    const int lrow  = tid >> 1;
    const int lhalf = tid & 1;

    constexpr int TOTAL = K1S + K2STEPS;

    f32x8 acc[2][2] = {};

    auto issue = [&](int s, int b) {
        if (s < K1S)
            stage_tiles(As[b], Ws[b], a1, LD1, mblk, W1, LD1, nblk, s * 32, lrow, lhalf);
        else
            stage_tiles(As[b], Ws[b], a2, KP, mblk, W2, KP, nblk, (s - K1S) * 32, lrow, lhalf);
    };

    issue(0, 0);
    for (int s = 0; s < TOTAL; ++s) {
        const int cur = s & 1;
        if (s + 1 < TOTAL) {
            issue(s + 1, cur ^ 1);          // stream next tiles while we compute
#if HAS_ASYNC_LDS
            __builtin_amdgcn_s_wait_asynccnt(4);   // our 4 ops for tile s have landed
#endif
        } else {
#if HAS_ASYNC_LDS
            __builtin_amdgcn_s_wait_asynccnt(0);
#endif
        }
        __syncthreads();

        FragU a[2], b[2];
        #pragma unroll
        for (int i = 0; i < 2; ++i) {
            const bf16* p = &As[cur][(wm * 32 + i * 16 + rsel) * LDA + c0];
            a[i].q[0] = *(const uint4*)p;
            a[i].q[1] = *(const uint4*)(p + 16);
        }
        #pragma unroll
        for (int j = 0; j < 2; ++j) {
            const bf16* p = &Ws[cur][(wn * 32 + j * 16 + rsel) * LDA + c0];
            b[j].q[0] = *(const uint4*)p;
            b[j].q[1] = *(const uint4*)(p + 16);
        }
        #pragma unroll
        for (int i = 0; i < 2; ++i)
            #pragma unroll
            for (int j = 0; j < 2; ++j)
                acc[i][j] = __builtin_amdgcn_wmma_f32_16x16x32_bf16(
                    false, a[i].v, false, b[j].v, (short)0, acc[i][j], false, false);
        __syncthreads();
    }

    // epilogue: bias + tanh + bf16 store (pad cols -> 0)
    // C/D layout: lane<16: N=lane, reg r -> M=r ; lane>=16: N=lane-16, reg r -> M=8+r
    #pragma unroll
    for (int j = 0; j < 2; ++j) {
        int n = nblk + wn * 32 + j * 16 + rsel;
        if (n >= KP) continue;
        bool valid = (n < HID);
        float bias = valid ? (b1[n] + b2[n]) : 0.0f;
        #pragma unroll
        for (int i = 0; i < 2; ++i) {
            #pragma unroll
            for (int r = 0; r < 8; ++r) {
                int m = mblk + wm * 32 + i * 16 + hi * 8 + r;
                float v = valid ? fast_tanh(acc[i][j][r] + bias) : 0.0f;
                out[(size_t)m * KP + n] = (bf16)v;
            }
        }
    }
}

// f32 [HID][K] -> bf16 [NP][Kpad], zero-filled padding in both dims.
__global__ void convert_pad_kernel(const float* __restrict__ src, bf16* __restrict__ dst,
                                   int K, int Kpad, int total)
{
    int idx = blockIdx.x * blockDim.x + threadIdx.x;
    if (idx >= total) return;
    int n = idx / Kpad, k = idx - n * Kpad;
    float v = (n < HID && k < K) ? src[(size_t)n * K + k] : 0.0f;
    dst[idx] = (bf16)v;
}

// xs f32 [T*B][47] -> xpad bf16 [T*B][64], zero-padded K.
__global__ void convert_x_kernel(const float* __restrict__ xs, bf16* __restrict__ xpad)
{
    int idx = blockIdx.x * blockDim.x + threadIdx.x;
    if (idx >= T_STEPS * BATCH * KP0) return;
    int r = idx >> 6, k = idx & 63;
    float v = (k < IN_DIM) ? xs[(size_t)r * IN_DIM + k] : 0.0f;
    xpad[idx] = (bf16)v;
}

__global__ void zero2_kernel(bf16* __restrict__ p0, bf16* __restrict__ p1, int total)
{
    int idx = blockIdx.x * blockDim.x + threadIdx.x;
    if (idx >= total) return;
    p0[idx] = (bf16)0.0f;
    p1[idx] = (bf16)0.0f;
}

// logits[b][o] = h1[b] . Wout[o] + bout[o]
__global__ void classifier_kernel(const bf16* __restrict__ h1, const float* __restrict__ Wout,
                                  const float* __restrict__ bout, float* __restrict__ out)
{
    int idx = blockIdx.x * blockDim.x + threadIdx.x;
    if (idx >= BATCH * OUT_DIM) return;
    int b = idx / OUT_DIM, o = idx - b * OUT_DIM;
    const bf16*  h = h1 + (size_t)b * KP;
    const float* w = Wout + (size_t)o * HID;
    float s = bout[o];
    for (int k = 0; k < HID; ++k) s += (float)h[k] * w[k];
    out[idx] = s;
}

extern "C" void kernel_launch(void* const* d_in, const int* in_sizes, int n_in,
                              void* d_out, int out_size, void* d_ws, size_t ws_size,
                              hipStream_t stream)
{
    (void)in_sizes; (void)n_in; (void)out_size; (void)ws_size;
    const float* xs   = (const float*)d_in[0];
    const float* Wih0 = (const float*)d_in[1];
    const float* Whh0 = (const float*)d_in[2];
    const float* bih0 = (const float*)d_in[3];
    const float* bhh0 = (const float*)d_in[4];
    const float* Wih1 = (const float*)d_in[5];
    const float* Whh1 = (const float*)d_in[6];
    const float* bih1 = (const float*)d_in[7];
    const float* bhh1 = (const float*)d_in[8];
    const float* Wout = (const float*)d_in[9];
    const float* bout = (const float*)d_in[10];
    float* out = (float*)d_out;

    char* ws = (char*)d_ws;
    auto al = [](size_t x) { return (x + 255) & ~(size_t)255; };
    size_t o = 0;
    bf16* xpad  = (bf16*)(ws + o); o += al((size_t)T_STEPS * BATCH * KP0 * 2);
    bf16* Wih0b = (bf16*)(ws + o); o += al((size_t)NP * KP0 * 2);
    bf16* Whh0b = (bf16*)(ws + o); o += al((size_t)NP * KP  * 2);
    bf16* Wih1b = (bf16*)(ws + o); o += al((size_t)NP * KP  * 2);
    bf16* Whh1b = (bf16*)(ws + o); o += al((size_t)NP * KP  * 2);
    bf16* H0[2]; bf16* H1[2];
    for (int i = 0; i < 2; ++i) { H0[i] = (bf16*)(ws + o); o += al((size_t)BATCH * KP * 2); }
    for (int i = 0; i < 2; ++i) { H1[i] = (bf16*)(ws + o); o += al((size_t)BATCH * KP * 2); }

    // per-launch prep (deterministic; inputs never mutated)
    {
        int nx = T_STEPS * BATCH * KP0;
        convert_x_kernel<<<(nx + 255) / 256, 256, 0, stream>>>(xs, xpad);
        int n0 = NP * KP0;
        convert_pad_kernel<<<(n0 + 255) / 256, 256, 0, stream>>>(Wih0, Wih0b, IN_DIM, KP0, n0);
        int n1 = NP * KP;
        convert_pad_kernel<<<(n1 + 255) / 256, 256, 0, stream>>>(Whh0, Whh0b, HID, KP, n1);
        convert_pad_kernel<<<(n1 + 255) / 256, 256, 0, stream>>>(Wih1, Wih1b, HID, KP, n1);
        convert_pad_kernel<<<(n1 + 255) / 256, 256, 0, stream>>>(Whh1, Whh1b, HID, KP, n1);
        int nh = BATCH * KP;
        zero2_kernel<<<(nh + 255) / 256, 256, 0, stream>>>(H0[0], H1[0], nh);
    }

    dim3 grid(11, 16);  // 11 N-tiles x 16 M-tiles = 176 blocks/step
    for (int t = 0; t < T_STEPS; ++t) {
        int p = t & 1, c = p ^ 1;
        const bf16* x_t = xpad + (size_t)t * BATCH * KP0;
        // layer 0: tanh(x_t Wih0^T + h0 Whh0^T + b)
        rnn_cell_kernel<KP0 / 32, KP0><<<grid, 128, 0, stream>>>(
            x_t, Wih0b, H0[p], Whh0b, bih0, bhh0, H0[c]);
        // layer 1: tanh(h0_new Wih1^T + h1 Whh1^T + b)
        rnn_cell_kernel<K2STEPS, KP><<<grid, 128, 0, stream>>>(
            H0[c], Wih1b, H1[p], Whh1b, bih1, bhh1, H1[c]);
    }
    // final write index for even T is buffer 0
    classifier_kernel<<<(BATCH * OUT_DIM + 255) / 256, 256, 0, stream>>>(H1[0], Wout, bout, out);
}
